// GraphSAGENet_66726611911375
// MI455X (gfx1250) — matrix operand order repaired and checked by reading, not compiled, not run on hardware
//
#include <hip/hip_runtime.h>
#include <hip/hip_bf16.h>

// GraphSAGE 2-layer on gfx1250: f32 atomic scatter-agg + bf16 WMMA GEMMs.
typedef __attribute__((ext_vector_type(16))) __bf16 v16bf;
typedef __attribute__((ext_vector_type(8)))  float  v8f;

#define NND 100000          // num nodes
#define DF  128             // feature dim (in = hidden = out)
#define NT  (NND / 16)      // 6250 node tiles, exact

__device__ __forceinline__ unsigned short f32_to_bf16_bits(float f) {
    unsigned int u = __float_as_uint(f);
    u += 0x7FFFu + ((u >> 16) & 1u);   // round-to-nearest-even
    return (unsigned short)(u >> 16);
}

// ---------------- utility kernels ----------------
__global__ void k_zero(float* __restrict__ p, long long n) {
    long long i = (long long)blockIdx.x * blockDim.x + threadIdx.x;
    long long stride = (long long)gridDim.x * blockDim.x;
    for (; i < n; i += stride) p[i] = 0.0f;
}

__global__ void k_deg(const int* __restrict__ row, float* __restrict__ deg, int E) {
    int e = blockIdx.x * blockDim.x + threadIdx.x;
    if (e < E) atomicAdd(&deg[row[e]], 1.0f);
}

__global__ void k_inv(const float* __restrict__ deg, float* __restrict__ inv, int n) {
    int i = blockIdx.x * blockDim.x + threadIdx.x;
    if (i < n) {
        float d = deg[i];
        inv[i] = (d > 0.0f) ? (1.0f / d) : 0.0f;
    }
}

// Pack one 128x128 f32 weight (nn.Linear layout W[o][k]) into a dense
// bf16 B-fragment order (32x16 per k-step, contiguous k matching A) so the GEMM B-load is a single
// contiguous 32B v16bf per lane.
// index i = (((ot*4 + ks)*32 + lane)*16 + e) ; o = ot*16 + (lane&15)
// k = ks*32 + 16*(lane>>4) + e
__global__ void k_pack(const float* __restrict__ W, unsigned short* __restrict__ P) {
    int i = blockIdx.x * blockDim.x + threadIdx.x;   // 0..16383
    if (i >= 8 * 4 * 32 * 16) return;
    int e    = i & 15;
    int lane = (i >> 4) & 31;
    int ksot = i >> 9;          // ot*4 + ks
    int ks   = ksot & 3;
    int ot   = ksot >> 2;
    int o  = ot * 16 + (lane & 15);
    int kh = lane >> 4;
    int k  = ks * 32 + kh * 16 + e;
    P[i] = f32_to_bf16_bits(W[o * DF + k]);
}

// One wave-group of 32 lanes per edge: gather 128 feats of src node (float4
// per lane, coalesced) and atomically scatter-add into the dst row.
__global__ void k_scatter(const float* __restrict__ feat,
                          const int* __restrict__ row, const int* __restrict__ col,
                          float* __restrict__ agg, int E) {
    long long t = (long long)blockIdx.x * blockDim.x + threadIdx.x;
    int e = (int)(t >> 5);
    if (e >= E) return;
    int lane = (int)(t & 31);
    int r = row[e];
    int c = col[e];
    const float4 v = *(const float4*)(feat + (long long)c * DF + lane * 4);
    float* dst = agg + (long long)r * DF + lane * 4;
    atomicAdd(dst + 0, v.x);
    atomicAdd(dst + 1, v.y);
    atomicAdd(dst + 2, v.z);
    atomicAdd(dst + 3, v.w);
}

// Convert 8 consecutive f32 (scaled by s) into bf16 elements [base, base+8) of
// a v16bf A fragment. Native __bf16 cast -> backend picks v_cvt_*bf16_f32 (RNE).
__device__ __forceinline__ void cvt8(v16bf& d, int base, const float* __restrict__ src, float s) {
    float4 a = *(const float4*)(src);
    float4 b = *(const float4*)(src + 4);
    d[base + 0] = (__bf16)(a.x * s);
    d[base + 1] = (__bf16)(a.y * s);
    d[base + 2] = (__bf16)(a.z * s);
    d[base + 3] = (__bf16)(a.w * s);
    d[base + 4] = (__bf16)(b.x * s);
    d[base + 5] = (__bf16)(b.y * s);
    d[base + 6] = (__bf16)(b.z * s);
    d[base + 7] = (__bf16)(b.w * s);
}

// out[16 nodes x 128 feats] = act( (agg*inv_deg) @ Wn^T + root @ Wr^T )
// One wave per 16-node tile; A kept in registers for full K=128; 8 output
// tiles x 4 k-steps x 2 matrices = 64 v_wmma_f32_16x16x32_bf16 per wave.
__global__ void __launch_bounds__(256)
k_gemm(const float* __restrict__ aggf, const float* __restrict__ rootf,
       const float* __restrict__ invdeg,
       const unsigned short* __restrict__ Pn, const unsigned short* __restrict__ Pr,
       float* __restrict__ out, int relu) {
    int wave = blockIdx.x * (blockDim.x >> 5) + (threadIdx.x >> 5);
    if (wave >= NT) return;                 // wave-uniform: EXEC stays all-1s
    int lane = threadIdx.x & 31;
    int m  = wave * 16 + (lane & 15);       // A-matrix row (same for both halves)
    int kh = lane >> 4;                     // K-half select
    float s = invdeg[m];

    const float* aggRow  = aggf  + (long long)m * DF;
    const float* rootRow = rootf + (long long)m * DF;

    // Load A fragments (contiguous K order, matching packed B): elem e <- K=kb+16*kh+e.
    // One contiguous 16-float run per lane, loaded as two 8-float halves.
    v16bf an[4], ar[4];
#pragma unroll
    for (int ks = 0; ks < 4; ++ks) {
        int b0 = ks * 32 + 16 * kh;
        int b1 = ks * 32 + 16 * kh + 8;
        cvt8(an[ks], 0, aggRow + b0, s);
        cvt8(an[ks], 8, aggRow + b1, s);
        cvt8(ar[ks], 0, rootRow + b0, 1.0f);
        cvt8(ar[ks], 8, rootRow + b1, 1.0f);
    }

    int ncol = lane & 15;
    for (int ot = 0; ot < 8; ++ot) {
        v8f c = {};
#pragma unroll
        for (int ks = 0; ks < 4; ++ks) {
            long long bidx = ((long long)(ot * 4 + ks) * 32 + lane) * 16;
            v16bf bn = *(const v16bf*)(Pn + bidx);
            v16bf br = *(const v16bf*)(Pr + bidx);
            c = __builtin_amdgcn_wmma_f32_16x16x32_bf16(false, an[ks], false, bn,
                                                        (short)0, c, false, false);
            c = __builtin_amdgcn_wmma_f32_16x16x32_bf16(false, ar[ks], false, br,
                                                        (short)0, c, false, false);
        }
        int colo = ot * 16 + ncol;
#pragma unroll
        for (int r = 0; r < 8; ++r) {
            float v = c[r];
            if (relu) v = fmaxf(v, 0.0f);
            out[(long long)(wave * 16 + r + 8 * kh) * DF + colo] = v;
        }
    }
}

extern "C" void kernel_launch(void* const* d_in, const int* in_sizes, int n_in,
                              void* d_out, int out_size, void* d_ws, size_t ws_size,
                              hipStream_t stream) {
    (void)n_in; (void)out_size; (void)ws_size;
    const float* x    = (const float*)d_in[0];
    const float* wn1  = (const float*)d_in[1];
    const float* wr1  = (const float*)d_in[2];
    const float* wn2  = (const float*)d_in[3];
    const float* wr2  = (const float*)d_in[4];
    const int*   eidx = (const int*)d_in[5];
    int E = in_sizes[5] / 2;
    const int* row = eidx;       // destinations
    const int* col = eidx + E;   // sources
    float* out = (float*)d_out;

    // workspace carve-out (~104 MB)
    size_t off = 0;
    char* ws = (char*)d_ws;
    auto take = [&](size_t bytes) -> void* {
        void* p = ws + off;
        off += (bytes + 255) & ~(size_t)255;
        return p;
    };
    float* deg = (float*)take(sizeof(float) * NND);
    float* inv = (float*)take(sizeof(float) * NND);
    float* agg = (float*)take(sizeof(float) * (size_t)NND * DF);
    float* h   = (float*)take(sizeof(float) * (size_t)NND * DF);
    unsigned short* Pn1 = (unsigned short*)take(sizeof(unsigned short) * 16384);
    unsigned short* Pr1 = (unsigned short*)take(sizeof(unsigned short) * 16384);
    unsigned short* Pn2 = (unsigned short*)take(sizeof(unsigned short) * 16384);
    unsigned short* Pr2 = (unsigned short*)take(sizeof(unsigned short) * 16384);

    const int B = 256;
    long long featN = (long long)NND * DF;

    // pack weights into WMMA B layout (bf16)
    k_pack<<<16384 / B, B, 0, stream>>>(wn1, Pn1);
    k_pack<<<16384 / B, B, 0, stream>>>(wr1, Pr1);
    k_pack<<<16384 / B, B, 0, stream>>>(wn2, Pn2);
    k_pack<<<16384 / B, B, 0, stream>>>(wr2, Pr2);

    // degrees + inverse
    k_zero<<<1024, B, 0, stream>>>(deg, NND);
    k_deg<<<(E + B - 1) / B, B, 0, stream>>>(row, deg, E);
    k_inv<<<(NND + B - 1) / B, B, 0, stream>>>(deg, inv, NND);

    int scatterBlocks = (int)(((long long)E * 32 + B - 1) / B);

    // layer 1: h = relu( mean_agg(x) @ Wn1^T + x @ Wr1^T )
    k_zero<<<4096, B, 0, stream>>>(agg, featN);
    k_scatter<<<scatterBlocks, B, 0, stream>>>(x, row, col, agg, E);
    k_gemm<<<(NT + 7) / 8, B, 0, stream>>>(agg, x, inv, Pn1, Pr1, h, 1);

    // layer 2: out = mean_agg(h) @ Wn2^T + h @ Wr2^T
    k_zero<<<4096, B, 0, stream>>>(agg, featN);
    k_scatter<<<scatterBlocks, B, 0, stream>>>(h, row, col, agg, E);
    k_gemm<<<(NT + 7) / 8, B, 0, stream>>>(agg, h, inv, Pn2, Pr2, out, 0);
}
